// CAModel_84490596647155
// MI455X (gfx1250) — matrix operand order repaired and checked by reading, not compiled, not run on hardware
//
#include <hip/hip_runtime.h>

typedef __attribute__((ext_vector_type(16))) _Float16 v16h;
typedef __attribute__((ext_vector_type(8)))  float    v8f;

// Tile: 8 rows x 16 cols of pixels per workgroup; one wave per row strip.
// B=4, C=16, H=W=512, HID=128. Features per pixel: 48 (x, gx, gy) padded to K=64.

__global__ __launch_bounds__(256) void ca_step_kernel(
    const float* __restrict__ xin, const float* __restrict__ w0,
    const float* __restrict__ bias0, const float* __restrict__ w1,
    const float* __restrict__ ru, float* __restrict__ out)
{
    const int tid  = threadIdx.x;
    const int lane = tid & 31;
    const int wid  = tid >> 5;

    // tile decode: 32 tiles along W, 64 tiles along H, 4 images
    int g = blockIdx.x;
    const int bimg = g >> 11;            // / (64*32)
    g &= 2047;
    const int th = g >> 5;
    const int tw = g & 31;
    const int hbase = th * 8;
    const int wbase = tw * 16;

    __shared__ __attribute__((aligned(32))) float    xs[16][10][18];     // 11520 B halo
    __shared__ __attribute__((aligned(32))) _Float16 w0f[16][32][16];    // 16384 B: 8 n-tiles x 2 k-chunks
    __shared__ __attribute__((aligned(32))) _Float16 w1f[4][32][16];     //  4096 B: 4 k-chunks
    __shared__ __attribute__((aligned(32))) _Float16 af[8][2][32][16];   // 16384 B: A frags (features)
    __shared__ __attribute__((aligned(32))) _Float16 hf[8][32][16];      //  8192 B: A2 frag (one k-chunk)

    // ---- stage w0 into WMMA B-fragment layout (lanes 0-15: K=0..7,16..23; 16-31: K=8..15,24..31) ----
    for (int idx = tid; idx < 8192; idx += 256) {
        int i16  = idx & 15;
        int ln   = (idx >> 4) & 31;
        int frag = idx >> 9;                       // j = frag>>1, kchunk = frag&1
        int k    = ((frag & 1) << 5) + ((i16 >> 3) << 4) + ((ln >> 4) << 3) + (i16 & 7);
        int nn   = ((frag >> 1) << 4) + (ln & 15);
        w0f[frag][ln][i16] = (_Float16)((k < 48) ? w0[nn * 48 + k] : 0.f);
    }
    // ---- stage w1 B-fragments ----
    for (int idx = tid; idx < 2048; idx += 256) {
        int i16 = idx & 15;
        int ln  = (idx >> 4) & 31;
        int kc  = idx >> 9;                        // 0..3
        int k   = (kc << 5) + ((i16 >> 3) << 4) + ((ln >> 4) << 3) + (i16 & 7);
        int nn  = ln & 15;
        w1f[kc][ln][i16] = (_Float16)(w1[nn * 128 + k]);
    }
    // ---- stage x halo (10 x 18 per channel, zero-padded at image edges) ----
    for (int idx = tid; idx < 2880; idx += 256) {
        int col = idx % 18;
        int t   = idx / 18;
        int row = t % 10;
        int ch  = t / 10;
        int hg  = hbase + row - 1;
        int wg  = wbase + col - 1;
        float v = 0.f;
        if ((unsigned)hg < 512u && (unsigned)wg < 512u)
            v = xin[(((size_t)bimg * 16 + ch) * 512 + hg) * 512 + wg];
        xs[ch][row][col] = v;
    }
    __syncthreads();

    // ---- compute 48 features/pixel, write straight into A-fragment layout (K padded to 64) ----
    for (int idx = tid; idx < 8192; idx += 256) {
        int m   = idx & 15;
        int f   = (idx >> 4) & 63;
        int hl2 = idx >> 10;
        float v = 0.f;
        if (f < 48) {
            int ch  = f & 15;
            int grp = f >> 4;                       // 0=x, 1=gx, 2=gy
            int r = hl2 + 1, cc = m + 1;
            if (grp == 0) {
                v = xs[ch][r][cc];
            } else if (grp == 1) {
                v = (xs[ch][r-1][cc+1] - xs[ch][r-1][cc-1])
                  + 2.f * (xs[ch][r][cc+1] - xs[ch][r][cc-1])
                  + (xs[ch][r+1][cc+1] - xs[ch][r+1][cc-1]);
            } else {
                v = (xs[ch][r+1][cc-1] + 2.f * xs[ch][r+1][cc] + xs[ch][r+1][cc+1])
                  - (xs[ch][r-1][cc-1] + 2.f * xs[ch][r-1][cc] + xs[ch][r-1][cc+1]);
            }
        }
        int kc   = f >> 5;
        int kl   = f & 31;
        int ln2  = m + (((kl >> 3) & 1) << 4);
        int vg   = ((kl & 7) >> 1) + ((kl >> 4) << 2);
        af[hl2][kc][ln2][2 * vg + (kl & 1)] = (_Float16)v;
    }
    __syncthreads();

    // ---- per-wave GEMMs: [16x64]x[64x128] -> relu -> [16x128]x[128x16] ----
    const int hl    = wid;
    const int hglob = hbase + hl;
    const int n     = lane & 15;
    const int mh    = (lane >> 4) << 3;            // D rows: v + mh

    v16h a0 = *reinterpret_cast<const v16h*>(&af[hl][0][lane][0]);
    v16h a1 = *reinterpret_cast<const v16h*>(&af[hl][1][lane][0]);
    v8f  d  = {};

    for (int kc = 0; kc < 4; ++kc) {
        #pragma unroll
        for (int jj = 0; jj < 2; ++jj) {
            const int j = kc * 2 + jj;             // hidden n-tile 0..7
            v16h bfr0 = *reinterpret_cast<const v16h*>(&w0f[j * 2 + 0][lane][0]);
            v16h bfr1 = *reinterpret_cast<const v16h*>(&w0f[j * 2 + 1][lane][0]);
            v8f acc = {};
            acc = __builtin_amdgcn_wmma_f32_16x16x32_f16(false, a0, false, bfr0,
                                                         (short)0, acc, false, false);
            acc = __builtin_amdgcn_wmma_f32_16x16x32_f16(false, a1, false, bfr1,
                                                         (short)0, acc, false, false);
            const float bv = bias0[j * 16 + n];
            // scatter relu(h) into A2 fragment layout for hid k-chunk kc
            const int kl   = (jj << 4) + n;        // (16j+n) & 31
            const int vg   = ((kl & 7) >> 1) + ((kl >> 4) << 2);
            const int half = kl & 1;
            const int lsh  = ((kl >> 3) & 1) << 4;
            #pragma unroll
            for (int v = 0; v < 8; ++v) {
                float hv = acc[v] + bv;
                hv = hv > 0.f ? hv : 0.f;
                hf[hl][(mh + v) + lsh - ((mh + v) >= 16 ? 0 : 0)][2 * vg + half] = (_Float16)hv;
            }
        }
        __syncthreads();
        v16h a2 = *reinterpret_cast<const v16h*>(&hf[hl][lane][0]);
        v16h bw = *reinterpret_cast<const v16h*>(&w1f[kc][lane][0]);
        d = __builtin_amdgcn_wmma_f32_16x16x32_f16(false, a2, false, bw,
                                                   (short)0, d, false, false);
        __syncthreads();
    }

    // ---- masked residual write: lane holds channel n, rows mh..mh+7 ----
    const size_t base_hw = (size_t)hglob * 512 + wbase;
    const size_t xbase   = ((size_t)bimg * 16 + n) * 262144 + base_hw;
    const size_t rbase   = (size_t)bimg * 262144 + base_hw;
    #pragma unroll
    for (int v = 0; v < 8; ++v) {
        const int m = mh + v;
        const float mv = ru[rbase + m] > 0.5f ? 1.0f : 0.0f;
        out[xbase + m] = xin[xbase + m] + d[v] * mv;
    }
}

extern "C" void kernel_launch(void* const* d_in, const int* in_sizes, int n_in,
                              void* d_out, int out_size, void* d_ws, size_t ws_size,
                              hipStream_t stream) {
    const float* x  = (const float*)d_in[0];
    const float* w0 = (const float*)d_in[1];
    const float* b0 = (const float*)d_in[2];
    const float* w1 = (const float*)d_in[3];
    const float* ru = (const float*)d_in[4];
    float* out = (float*)d_out;
    // 4 images * (512/8) * (512/16) = 8192 workgroups, 256 threads (8 waves) each
    hipLaunchKernelGGL(ca_step_kernel, dim3(8192), dim3(256), 0, stream,
                       x, w0, b0, w1, ru, out);
}
